// VarianceAdapter_36000415875784
// MI455X (gfx1250) — matrix operand order repaired and checked by reading, not compiled, not run on hardware
//
#include <hip/hip_runtime.h>
#include <hip/hip_bf16.h>

// ---------------------------------------------------------------------------
// VarianceAdapter for MI455X (gfx1250): bf16 WMMA convs, gather-based path
// expansion, fused LayerNorm.  Layouts:
//   activations (ws):  [b][t][c] bf16   (t-major so conv tiles are contiguous)
//   weights (ws):      [p*l][k][co][ci] bf16
//   d_out: out[B,D,TY] f32 | pred_dur[B,TX] | pred_pitch[B,TY] | pred_energy[B,TY]
// ---------------------------------------------------------------------------

typedef __bf16 bf16;
typedef __attribute__((ext_vector_type(16))) __bf16 v16bf;
typedef __attribute__((ext_vector_type(8)))  float  v8f;
typedef __attribute__((ext_vector_type(4)))  int    v4i;

#define B_    16
#define TX_   512
#define TY_   2048
#define DCH   256
#define KW    3
#define NL    2
#define NP    3
#define EPS_  1e-5f

#define TILE_T     32
#define HS_STRIDE  264   // bf16 per LDS input row (t-major), padded vs bank conflicts
#define YS_STRIDE  260   // f32 per LDS output row

// Async global->LDS staging if the toolchain exposes the gfx1250 builtin;
// otherwise fall back to the (proven) sync copy through VGPRs.
#if defined(__has_builtin)
#if __has_builtin(__builtin_amdgcn_global_load_async_to_lds_b128)
#define HAVE_ASYNC_LDS 1
#endif
#endif
#ifndef HAVE_ASYNC_LDS
#define HAVE_ASYNC_LDS 0
#endif

#if HAVE_ASYNC_LDS
// param 0: pointer to v4i in addrspace(1) ("__device__"), param 1: addrspace(3)
typedef __attribute__((address_space(1))) v4i as1_v4i;
typedef __attribute__((address_space(3))) v4i as3_v4i;
#endif

union FragBF { v16bf v; unsigned int u[8]; uint4 q[2]; };

// ---------------------------------------------------------------------------
// conv1d(k=3,pad=1) + bias + ReLU + channel-LayerNorm + mask, bf16 in/out.
// One block: 256 co x 32 t tile for one batch. 8 waves, each 2x2 WMMA tiles.
// ---------------------------------------------------------------------------
__global__ __launch_bounds__(256) void conv_ln_kernel(
    const bf16* __restrict__ Hin,          // [B][T][DCH]
    const bf16* __restrict__ W,            // [3][DCH(co)][DCH(ci)]
    const float* __restrict__ bias,        // [DCH]
    const float* __restrict__ gamma,       // [DCH]
    const float* __restrict__ beta,        // [DCH]
    const float* __restrict__ mask,        // [B][T]
    bf16* __restrict__ Hout,               // [B][T][DCH]
    int T)
{
  __shared__ __align__(16) bf16 Hs[(TILE_T + 2) * HS_STRIDE];
  __shared__ float Ys[TILE_T * YS_STRIDE];
  __shared__ float mu_s[TILE_T], rs_s[TILE_T];

  const int b   = blockIdx.y;
  const int t0  = blockIdx.x * TILE_T;
  const int tid = threadIdx.x;
  const bf16* hb = Hin + (size_t)b * T * DCH;

  // Stage input rows t0-1 .. t0+TILE_T (34 rows) into LDS; zero halo OOB.
  for (int idx = tid; idx < (TILE_T + 2) * (DCH / 8); idx += 256) {
    int row = idx / (DCH / 8);
    int cg  = (idx % (DCH / 8)) * 8;                   // 8 bf16 = 16 bytes
    int t   = t0 - 1 + row;
    bf16* dst = &Hs[row * HS_STRIDE + cg];
#if HAVE_ASYNC_LDS
    if (t >= 0 && t < T) {
      __builtin_amdgcn_global_load_async_to_lds_b128(
          (as1_v4i*)const_cast<bf16*>(hb + (size_t)t * DCH + cg),
          (as3_v4i*)dst, 0, 0);
    } else {
      *(uint4*)dst = make_uint4(0u, 0u, 0u, 0u);
    }
#else
    uint4 val = make_uint4(0u, 0u, 0u, 0u);
    if (t >= 0 && t < T)
      val = *(const uint4*)(hb + (size_t)t * DCH + cg);
    *(uint4*)dst = val;
#endif
  }
#if HAVE_ASYNC_LDS
#if defined(__has_builtin) && __has_builtin(__builtin_amdgcn_s_wait_asynccnt)
  __builtin_amdgcn_s_wait_asynccnt(0);
#else
  asm volatile("s_wait_asynccnt 0x0" ::: "memory");
#endif
#endif
  __syncthreads();

  const int wave = tid >> 5;
  const int lane = tid & 31;
  const int half = lane >> 4;     // lane group 0..15 / 16..31
  const int mn   = lane & 15;     // A: row m (co), B: col n (t)

  v8f acc[2][2] = {};             // [co_sub][t_sub]

  for (int k = 0; k < KW; ++k) {
    for (int cc = 0; cc < DCH; cc += 32) {
      // A fragments (weights) from global/L2. A k order used here (contiguous):
      // lane(half,m): vgpr pair jj -> K = half*16 + jj*2 (matches B).
      FragBF A[2];
      #pragma unroll
      for (int cs = 0; cs < 2; ++cs) {
        int co = wave * 32 + cs * 16 + mn;
        const bf16* wp = W + ((size_t)(k * DCH + co)) * DCH + cc + half * 16;
        #pragma unroll
        for (int jj = 0; jj < 8; ++jj) {
          int Koff = jj * 2;
          A[cs].u[jj] = *(const unsigned int*)(wp + Koff);
        }
        // hide L2 latency for the next ci-chunk behind this WMMA group
        __builtin_prefetch((const void*)(wp + 32), 0, 3);
      }
      // B fragments (activations) from LDS. B k order used here (contiguous):
      // lane(half,n): 16 consecutive ci at K = half*16 + j  -> 2x b128 loads.
      #pragma unroll
      for (int ts = 0; ts < 2; ++ts) {
        FragBF Bf;
        int row = ts * 16 + mn + k;                       // +1 halo, -1 pad => +k
        const bf16* bp = &Hs[row * HS_STRIDE + cc + half * 16];
        Bf.q[0] = *(const uint4*)(bp);
        Bf.q[1] = *(const uint4*)(bp + 8);
        #pragma unroll
        for (int cs = 0; cs < 2; ++cs)
          acc[cs][ts] = __builtin_amdgcn_wmma_f32_16x16x32_bf16(
              false, A[cs].v, false, Bf.v, (short)0, acc[cs][ts], false, false);
      }
    }
  }

  // bias + ReLU -> LDS fp32 tile Ys[t][co] (C/D layout: M = v + half*8, N = mn)
  #pragma unroll
  for (int cs = 0; cs < 2; ++cs)
    #pragma unroll
    for (int ts = 0; ts < 2; ++ts)
      #pragma unroll
      for (int v = 0; v < 8; ++v) {
        int co = wave * 32 + cs * 16 + half * 8 + v;
        int tl = ts * 16 + mn;
        float y = acc[cs][ts][v] + bias[co];
        Ys[tl * YS_STRIDE + co] = fmaxf(y, 0.0f);
      }
  __syncthreads();

  // Channel LayerNorm stats: 8 lanes per t column, each sums 32 channels.
  {
    int t  = tid >> 3;
    int cg = (tid & 7) * 32;
    float s = 0.f, ss = 0.f;
    const float* yr = &Ys[t * YS_STRIDE + cg];
    #pragma unroll
    for (int i = 0; i < 32; ++i) { float v = yr[i]; s += v; ss += v * v; }
    #pragma unroll
    for (int off = 4; off > 0; off >>= 1) {
      s  += __shfl_down(s,  off, 8);
      ss += __shfl_down(ss, off, 8);
    }
    if ((tid & 7) == 0) {
      float mu  = s * (1.0f / DCH);
      float var = ss * (1.0f / DCH) - mu * mu;
      mu_s[t] = mu;
      rs_s[t] = rsqrtf(var + EPS_);
    }
  }
  __syncthreads();

  // normalize * gamma + beta, * mask -> bf16 global (coalesced over c)
  bf16* ob = Hout + (size_t)b * T * DCH;
  for (int idx = tid; idx < TILE_T * DCH; idx += 256) {
    int tl = idx >> 8;
    int c  = idx & (DCH - 1);
    int t  = t0 + tl;
    float v = Ys[tl * YS_STRIDE + c];
    float o = (v - mu_s[tl]) * rs_s[tl] * gamma[c] + beta[c];
    o *= mask[(size_t)b * T + t];
    ob[(size_t)t * DCH + c] = (bf16)o;
  }
}

// --------------------------- support kernels -------------------------------

// Wb[pl][k][co][ci] = conv_w[pl][co][ci][k], fp32 -> bf16
__global__ void cvt_w_kernel(const float* __restrict__ w, bf16* __restrict__ Wb)
{
  size_t i = (size_t)blockIdx.x * blockDim.x + threadIdx.x;
  size_t N = (size_t)NP * NL * KW * DCH * DCH;
  if (i >= N) return;
  int ci = (int)(i & (DCH - 1));
  int co = (int)((i >> 8) & (DCH - 1));
  int k  = (int)((i >> 16) % KW);
  int pl = (int)((i >> 16) / KW);
  Wb[i] = (bf16)w[(((size_t)pl * DCH + co) * DCH + ci) * KW + k];
}

// Xb[b][t][c] = (bf16) x[b][c][t]
__global__ void cvt_x_kernel(const float* __restrict__ x, bf16* __restrict__ Xb)
{
  size_t i = (size_t)blockIdx.x * blockDim.x + threadIdx.x;
  if (i >= (size_t)B_ * TX_ * DCH) return;
  int c = (int)(i & (DCH - 1));
  int t = (int)((i >> 8) & (TX_ - 1));
  int b = (int)(i >> 17);
  Xb[i] = (bf16)x[((size_t)b * DCH + c) * TX_ + t];
}

// per-batch inclusive scan over TX durations
__global__ void cumsum_kernel(const float* __restrict__ dur, float* __restrict__ cum)
{
  __shared__ float s[TX_];
  int b = blockIdx.x, t = threadIdx.x;
  s[t] = dur[b * TX_ + t];
  __syncthreads();
  for (int off = 1; off < TX_; off <<= 1) {
    float v = (t >= off) ? s[t - off] : 0.f;
    __syncthreads();
    s[t] += v;
    __syncthreads();
  }
  cum[b * TX_ + t] = s[t];
}

// mask_x[b][x] = max_y x_mask ; coalesced row sweeps
__global__ void maskx_kernel(const float* __restrict__ xm, float* __restrict__ mask_x)
{
  int b = blockIdx.x, x = threadIdx.x;
  const float* p = xm + (size_t)b * TY_ * TX_ + x;
  float m = 0.f;
  for (int y = 0; y < TY_; ++y) m = fmaxf(m, p[(size_t)y * TX_]);
  mask_x[b * TX_ + x] = m;
}

// mask_y[b][y] = max_x x_mask
__global__ void masky_kernel(const float* __restrict__ xm, float* __restrict__ mask_y)
{
  int y = blockIdx.x * blockDim.x + threadIdx.x;
  int b = blockIdx.y;
  const float* p = xm + ((size_t)b * TY_ + y) * TX_;
  float m = 0.f;
  for (int x = 0; x < TX_; ++x) m = fmaxf(m, p[x]);
  mask_y[b * TY_ + y] = m;
}

// path expansion as gather (path rows are one-hot) + fused pitch/energy embed.
// Writes final `out` (fp32, [b][c][y]) and bf16 xe ([b][y][c]) for predictors.
__global__ void expand_kernel(const float* __restrict__ x, const float* __restrict__ cum,
                              const float* __restrict__ pitch, const float* __restrict__ energy,
                              const float* __restrict__ pw, const float* __restrict__ pb,
                              const float* __restrict__ ew, const float* __restrict__ eb,
                              float* __restrict__ out, bf16* __restrict__ Xeb)
{
  int y = blockIdx.x * blockDim.x + threadIdx.x;
  int b = blockIdx.y;
  const float* cb = cum + b * TX_;
  float total = cb[TX_ - 1];
  float fy = (float)y;
  int lo = 0, hi = TX_;                       // first x with cum[x] > y
  while (lo < hi) { int mid = (lo + hi) >> 1; if (fy < cb[mid]) hi = mid; else lo = mid + 1; }
  int xi = lo;
  bool valid = (fy < total) && (xi < TX_);
  float pv = pitch[b * TY_ + y];
  float ev = energy[b * TY_ + y];
  const float* xb = x + (size_t)b * DCH * TX_;
  float* ob = out + (size_t)b * DCH * TY_;
  bf16* xr = Xeb + ((size_t)b * TY_ + y) * DCH;
  for (int c = 0; c < DCH; ++c) {
    float v = valid ? xb[(size_t)c * TX_ + xi] : 0.0f;
    ob[(size_t)c * TY_ + y] = v + pw[c] * pv + pb[c] + ew[c] * ev + eb[c];
    xr[c] = (bf16)v;
  }
}

// 1x1 projection D->1 (+bias, *mask)
__global__ void proj_kernel(const bf16* __restrict__ H, const float* __restrict__ pw,
                            const float* __restrict__ pbias, const float* __restrict__ mask,
                            float* __restrict__ out, int T)
{
  int t = blockIdx.x * blockDim.x + threadIdx.x;
  int b = blockIdx.y;
  if (t >= T) return;
  const bf16* hp = H + ((size_t)b * T + t) * DCH;
  float s = 0.f;
  for (int c = 0; c < DCH; ++c) s += (float)hp[c] * pw[c];
  out[(size_t)b * T + t] = (s + pbias[0]) * mask[(size_t)b * T + t];
}

// ---------------------------------------------------------------------------

extern "C" void kernel_launch(void* const* d_in, const int* in_sizes, int n_in,
                              void* d_out, int out_size, void* d_ws, size_t ws_size,
                              hipStream_t stream) {
  (void)in_sizes; (void)n_in; (void)out_size; (void)ws_size;

  const float* x        = (const float*)d_in[0];
  const float* x_mask   = (const float*)d_in[1];
  const float* duration = (const float*)d_in[2];
  const float* pitch    = (const float*)d_in[3];
  const float* energy   = (const float*)d_in[4];
  const float* conv_w   = (const float*)d_in[5];
  const float* conv_b   = (const float*)d_in[6];
  const float* ln_g     = (const float*)d_in[7];
  const float* ln_b     = (const float*)d_in[8];
  const float* proj_w   = (const float*)d_in[9];
  const float* proj_b   = (const float*)d_in[10];
  const float* pe_w     = (const float*)d_in[11];
  const float* pe_b     = (const float*)d_in[12];
  const float* ee_w     = (const float*)d_in[13];
  const float* ee_b     = (const float*)d_in[14];

  // d_out regions (reference return order)
  float* out_xe   = (float*)d_out;                         // [B,D,TY]
  float* pred_dur = out_xe + (size_t)B_ * DCH * TY_;       // [B,TX]
  float* pred_pit = pred_dur + (size_t)B_ * TX_;           // [B,TY]
  float* pred_en  = pred_pit + (size_t)B_ * TY_;           // [B,TY]

  // workspace carve-up (all offsets 256B aligned)
  char* ws = (char*)d_ws;
  size_t off = 0;
  float* cum    = (float*)(ws + off); off += (size_t)B_ * TX_ * 4;
  float* mask_x = (float*)(ws + off); off += (size_t)B_ * TX_ * 4;
  float* mask_y = (float*)(ws + off); off += (size_t)B_ * TY_ * 4;
  bf16*  Wb     = (bf16*)(ws + off);  off += (size_t)NP * NL * KW * DCH * DCH * 2;
  bf16*  Xb     = (bf16*)(ws + off);  off += (size_t)B_ * TX_ * DCH * 2;
  bf16*  Xeb    = (bf16*)(ws + off);  off += (size_t)B_ * TY_ * DCH * 2;
  bf16*  H1     = (bf16*)(ws + off);  off += (size_t)B_ * TY_ * DCH * 2;
  bf16*  H2     = (bf16*)(ws + off);  off += (size_t)B_ * TY_ * DCH * 2;

  // --- prep ---
  {
    size_t Nw = (size_t)NP * NL * KW * DCH * DCH;
    cvt_w_kernel<<<dim3((unsigned)((Nw + 255) / 256)), 256, 0, stream>>>(conv_w, Wb);
    size_t Nx = (size_t)B_ * TX_ * DCH;
    cvt_x_kernel<<<dim3((unsigned)((Nx + 255) / 256)), 256, 0, stream>>>(x, Xb);
    cumsum_kernel<<<B_, TX_, 0, stream>>>(duration, cum);
    maskx_kernel<<<B_, TX_, 0, stream>>>(x_mask, mask_x);
    masky_kernel<<<dim3(TY_ / 256, B_), 256, 0, stream>>>(x_mask, mask_y);
    expand_kernel<<<dim3(TY_ / 256, B_), 256, 0, stream>>>(
        x, cum, pitch, energy, pe_w, pe_b, ee_w, ee_b, out_xe, Xeb);
  }

  // --- three variance predictors ---
  const size_t WPL = (size_t)KW * DCH * DCH;  // bf16 elems per (p,l)
  for (int p = 0; p < NP; ++p) {
    const bf16* hin = (p == 0) ? Xb : Xeb;
    int T = (p == 0) ? TX_ : TY_;
    const float* msk = (p == 0) ? mask_x : mask_y;
    dim3 cgrid(T / TILE_T, B_);

    conv_ln_kernel<<<cgrid, 256, 0, stream>>>(
        hin, Wb + (size_t)(p * NL + 0) * WPL,
        conv_b + (size_t)(p * NL + 0) * DCH, ln_g + (size_t)(p * NL + 0) * DCH,
        ln_b + (size_t)(p * NL + 0) * DCH, msk, H1, T);
    conv_ln_kernel<<<cgrid, 256, 0, stream>>>(
        H1, Wb + (size_t)(p * NL + 1) * WPL,
        conv_b + (size_t)(p * NL + 1) * DCH, ln_g + (size_t)(p * NL + 1) * DCH,
        ln_b + (size_t)(p * NL + 1) * DCH, msk, H2, T);

    float* pred = (p == 0) ? pred_dur : (p == 1) ? pred_pit : pred_en;
    proj_kernel<<<dim3(T / 256, B_), 256, 0, stream>>>(
        H2, proj_w + (size_t)p * DCH, proj_b + p, msk, pred, T);
  }
}
